// Attention_45131516346540
// MI455X (gfx1250) — matrix operand order repaired and checked by reading, not compiled, not run on hardware
//
#include <hip/hip_runtime.h>
#include <hip/hip_bf16.h>

// ---------------------------------------------------------------------------
// PVT spatial-reduction attention, fully WMMA-based for gfx1250 (wave32).
// B=4, N=4096, C=512, H=8, hd=64, SR=2 -> N'=1024.
// All matmuls (q, conv-as-GEMM, kv, qk^T, PV, proj) use v_wmma_f32_16x16x32_f16.
// Attention is flash-style (online softmax) with K/V chunks staged into LDS
// via GLOBAL_LOAD_ASYNC_TO_LDS_B128 (ASYNCcnt) shared by all 8 waves.
// ---------------------------------------------------------------------------

typedef __attribute__((ext_vector_type(16))) _Float16 v16h;
typedef __attribute__((ext_vector_type(8)))  _Float16 h8;
typedef __attribute__((ext_vector_type(8)))  float    v8f;
typedef __attribute__((ext_vector_type(4)))  int      i4vec;
typedef __attribute__((address_space(1))) i4vec* gbl_i4p;   // global int4*
typedef __attribute__((address_space(3))) i4vec* lds_i4p;   // LDS int4*

#define WMMA_F16(a, b, c) \
  __builtin_amdgcn_wmma_f32_16x16x32_f16(false, (a), false, (b), (short)0, (c), false, false)

constexpr int BN    = 4;
constexpr int NTOK  = 4096;   // tokens per batch (64x64)
constexpr int CDIM  = 512;
constexpr int HN    = 8;
constexpr int HD    = 64;
constexpr int NP    = 1024;   // reduced tokens (32x32)
constexpr int KCONV = 2048;   // 4 * 512 conv contraction
constexpr float ATT_SCALE = 0.125f;  // 1/sqrt(64)
constexpr float LNEPS = 1e-5f;

// ---- CDNA5 async copy (global -> LDS, 16B per lane, tracked by ASYNCcnt) --
__device__ __forceinline__ void async_copy_16B(const _Float16* g, _Float16* l) {
#if __has_builtin(__builtin_amdgcn_global_load_async_to_lds_b128)
  __builtin_amdgcn_global_load_async_to_lds_b128((gbl_i4p)g, (lds_i4p)l, 0, 0);
#else
  *(h8*)l = *(const h8*)g;   // synchronous fallback
#endif
}

__device__ __forceinline__ void wait_async_lds() {
#if __has_builtin(__builtin_amdgcn_s_wait_asynccnt)
  __builtin_amdgcn_s_wait_asynccnt(0);
#else
  asm volatile("s_wait_asynccnt 0x0" ::: "memory");
#endif
}

// ---- fragment loaders -----------------------------------------------------
// A fragment 16x32 f16 (rows = M, row-major source with leading dim lda):
//   lane l (half = l>>4, r = l&15): row0+r, cols k0+8*half..+7 and k0+16+8*half..+7
__device__ __forceinline__ v16h load_a_frag(const _Float16* A, int row0, int k0, int lda) {
  const int l = threadIdx.x & 31, half = l >> 4, r = l & 15;
  const _Float16* p = A + (size_t)(row0 + r) * lda + (k0 + 8 * half);
  h8 lo = *(const h8*)p;
  h8 hi = *(const h8*)(p + 16);
  v16h f;
#pragma unroll
  for (int i = 0; i < 8; ++i) { f[i] = lo[i]; f[i + 8] = hi[i]; }
  return f;
}

// B fragment 32x16 f16 from Bt[n][k] row-major (Bt = weight in (out,in) layout):
//   lane l: n = n0 + (l&15), cols k0+8*half..+7 and k0+16+8*half..+7
__device__ __forceinline__ v16h load_b_frag(const _Float16* Bt, int n0, int k0, int ldb) {
  const int l = threadIdx.x & 31, half = l >> 4, r = l & 15;
  const _Float16* p = Bt + (size_t)(n0 + r) * ldb + (k0 + 8 * half);
  h8 lo = *(const h8*)p;
  h8 hi = *(const h8*)(p + 16);
  v16h f;
#pragma unroll
  for (int i = 0; i < 8; ++i) { f[i] = lo[i]; f[i + 8] = hi[i]; }
  return f;
}

// ---- elementwise helpers --------------------------------------------------
__global__ void k_f32_to_f16(const float* __restrict__ in, _Float16* __restrict__ out, int n) {
  int i = blockIdx.x * blockDim.x + threadIdx.x;
  if (i < n) out[i] = (_Float16)in[i];
}

// pack sr_w (OIHW, [512][512][2][2]) into Wc[co][kidx*512+ci], kidx = kh*2+kw
__global__ void k_pack_srw(const float* __restrict__ srw, _Float16* __restrict__ wc) {
  int i = blockIdx.x * blockDim.x + threadIdx.x;
  if (i >= CDIM * KCONV) return;
  int co = i >> 11, kk = i & 2047;
  int kidx = kk >> 9, ci = kk & 511;
  int kh = kidx >> 1, kw = kidx & 1;
  wc[i] = (_Float16)srw[(((size_t)co * CDIM + ci) * 2 + kh) * 2 + kw];
}

// transpose v half of kv: vT[b*512 + c][m] = kvh[(b*NP+m)][512 + c]
__global__ void k_transpose_v(const _Float16* __restrict__ kvh, _Float16* __restrict__ vT) {
  int i = blockIdx.x * blockDim.x + threadIdx.x;
  if (i >= BN * CDIM * NP) return;
  int rT = i >> 10, m = i & 1023;
  int b = rT >> 9, c = rT & 511;
  vT[i] = kvh[((size_t)(b * NP + m)) * 1024 + 512 + c];
}

// ---- generic WMMA GEMM: C = A * Bt^T, f16 out -----------------------------
// block = 256 thr (8 waves, 4x2), block tile 128x64, wave tile 32x32
__global__ void k_gemm_f16out(const _Float16* __restrict__ A, const _Float16* __restrict__ Bt,
                              _Float16* __restrict__ Co, int K, int ldc) {
  const int wave = threadIdx.x >> 5;
  const int r0 = blockIdx.y * 128 + (wave >> 1) * 32;
  const int c0 = blockIdx.x * 64 + (wave & 1) * 32;
  v8f acc[2][2] = {};
  for (int k0 = 0; k0 < K; k0 += 32) {
    __builtin_prefetch(A + (size_t)r0 * K + k0 + 128, 0, 0);
    v16h a0 = load_a_frag(A, r0, k0, K);
    v16h a1 = load_a_frag(A, r0 + 16, k0, K);
    v16h b0 = load_b_frag(Bt, c0, k0, K);
    v16h b1 = load_b_frag(Bt, c0 + 16, k0, K);
    acc[0][0] = WMMA_F16(a0, b0, acc[0][0]);
    acc[0][1] = WMMA_F16(a0, b1, acc[0][1]);
    acc[1][0] = WMMA_F16(a1, b0, acc[1][0]);
    acc[1][1] = WMMA_F16(a1, b1, acc[1][1]);
  }
  const int l = threadIdx.x & 31, half = l >> 4, cn = l & 15;
#pragma unroll
  for (int i = 0; i < 2; ++i)
#pragma unroll
    for (int j = 0; j < 2; ++j)
#pragma unroll
      for (int e = 0; e < 8; ++e)
        Co[(size_t)(r0 + i * 16 + e + 8 * half) * ldc + c0 + j * 16 + cn] = (_Float16)acc[i][j][e];
}

// same but f32 out + bias (final projection -> d_out)
__global__ void k_gemm_f32out_bias(const _Float16* __restrict__ A, const _Float16* __restrict__ Bt,
                                   const float* __restrict__ bias, float* __restrict__ Co,
                                   int K, int ldc) {
  const int wave = threadIdx.x >> 5;
  const int r0 = blockIdx.y * 128 + (wave >> 1) * 32;
  const int c0 = blockIdx.x * 64 + (wave & 1) * 32;
  v8f acc[2][2] = {};
  for (int k0 = 0; k0 < K; k0 += 32) {
    v16h a0 = load_a_frag(A, r0, k0, K);
    v16h a1 = load_a_frag(A, r0 + 16, k0, K);
    v16h b0 = load_b_frag(Bt, c0, k0, K);
    v16h b1 = load_b_frag(Bt, c0 + 16, k0, K);
    acc[0][0] = WMMA_F16(a0, b0, acc[0][0]);
    acc[0][1] = WMMA_F16(a0, b1, acc[0][1]);
    acc[1][0] = WMMA_F16(a1, b0, acc[1][0]);
    acc[1][1] = WMMA_F16(a1, b1, acc[1][1]);
  }
  const int l = threadIdx.x & 31, half = l >> 4, cn = l & 15;
#pragma unroll
  for (int i = 0; i < 2; ++i)
#pragma unroll
    for (int j = 0; j < 2; ++j) {
      float bz = bias[c0 + j * 16 + cn];
#pragma unroll
      for (int e = 0; e < 8; ++e)
        Co[(size_t)(r0 + i * 16 + e + 8 * half) * ldc + c0 + j * 16 + cn] = acc[i][j][e] + bz;
    }
}

// ---- SR conv(2x2, stride 2) as a gathered-A GEMM, K = 2048 ----------------
__device__ __forceinline__ v16h load_a_conv(const _Float16* xh, int r0, int k0) {
  const int l = threadIdx.x & 31, half = l >> 4, r = l & 15;
  int row = r0 + r;                 // conv-output row in [0, B*NP)
  int b = row >> 10, mm = row & 1023;
  int oh_ = mm >> 5, ow_ = mm & 31;
  int kidx = k0 >> 9;               // which of the 4 conv taps
  int ci0 = (k0 & 511) + 8 * half;
  int kh = kidx >> 1, kw = kidx & 1;
  int srow = b * NTOK + (2 * oh_ + kh) * 64 + (2 * ow_ + kw);
  const _Float16* p = xh + (size_t)srow * CDIM + ci0;
  h8 lo = *(const h8*)p;
  h8 hi = *(const h8*)(p + 16);
  v16h f;
#pragma unroll
  for (int i = 0; i < 8; ++i) { f[i] = lo[i]; f[i + 8] = hi[i]; }
  return f;
}

__global__ void k_conv_gemm(const _Float16* __restrict__ xh, const _Float16* __restrict__ Wc,
                            const float* __restrict__ srb, float* __restrict__ convo) {
  const int wave = threadIdx.x >> 5;
  const int r0 = blockIdx.y * 128 + (wave >> 1) * 32;
  const int c0 = blockIdx.x * 64 + (wave & 1) * 32;
  v8f acc[2][2] = {};
  for (int k0 = 0; k0 < KCONV; k0 += 32) {
    v16h a0 = load_a_conv(xh, r0, k0);
    v16h a1 = load_a_conv(xh, r0 + 16, k0);
    v16h b0 = load_b_frag(Wc, c0, k0, KCONV);
    v16h b1 = load_b_frag(Wc, c0 + 16, k0, KCONV);
    acc[0][0] = WMMA_F16(a0, b0, acc[0][0]);
    acc[0][1] = WMMA_F16(a0, b1, acc[0][1]);
    acc[1][0] = WMMA_F16(a1, b0, acc[1][0]);
    acc[1][1] = WMMA_F16(a1, b1, acc[1][1]);
  }
  const int l = threadIdx.x & 31, half = l >> 4, cn = l & 15;
#pragma unroll
  for (int i = 0; i < 2; ++i)
#pragma unroll
    for (int j = 0; j < 2; ++j) {
      float bz = srb[c0 + j * 16 + cn];
#pragma unroll
      for (int e = 0; e < 8; ++e)
        convo[(size_t)(r0 + i * 16 + e + 8 * half) * CDIM + c0 + j * 16 + cn] = acc[i][j][e] + bz;
    }
}

// ---- LayerNorm over C=512 (one wave per token row), f32 in -> f16 out -----
__global__ void k_layernorm(const float* __restrict__ in, const float* __restrict__ g,
                            const float* __restrict__ bt, _Float16* __restrict__ out) {
  const int wave = threadIdx.x >> 5, lane = threadIdx.x & 31;
  const int row = blockIdx.x * 8 + wave;
  const float* rp = in + (size_t)row * CDIM;
  float vv[16], sum = 0.f, ss = 0.f;
#pragma unroll
  for (int j = 0; j < 16; ++j) {
    float v = rp[lane + 32 * j];
    vv[j] = v; sum += v; ss += v * v;
  }
#pragma unroll
  for (int m = 1; m < 32; m <<= 1) {
    sum += __shfl_xor(sum, m, 32);
    ss  += __shfl_xor(ss, m, 32);
  }
  float mu = sum * (1.0f / CDIM);
  float var = ss * (1.0f / CDIM) - mu * mu;
  float rs = rsqrtf(var + LNEPS);
  _Float16* op = out + (size_t)row * CDIM;
#pragma unroll
  for (int j = 0; j < 16; ++j) {
    int c = lane + 32 * j;
    op[c] = (_Float16)((vv[j] - mu) * rs * g[c] + bt[c]);
  }
}

// ---- flash attention ------------------------------------------------------
// block = 8 waves (256 thr), 128 q rows / block, 64-key chunks.
// K/V chunks staged once per block into LDS with async-to-LDS copies and
// shared by all waves; P re-fragmented through a per-wave LDS strip.
// grid: (32 q-tiles, 32 b*h)
__global__ void k_flash_attn(const _Float16* __restrict__ qh, const _Float16* __restrict__ kvh,
                             const _Float16* __restrict__ vT, _Float16* __restrict__ oh) {
  __shared__ _Float16 Kb[64 * 64];       // K chunk: [key][d]    (8 KB)
  __shared__ _Float16 Vb[64 * 64];       // V chunk: [d][key]    (8 KB)
  __shared__ _Float16 Pl[8 * 16 * 72];   // per-wave 16x64 P strip (18 KB)
  const int tid = threadIdx.x;
  const int wave = tid >> 5;
  const int l = tid & 31, half = l >> 4, cn = l & 15;
  const int bh = blockIdx.y, b = bh >> 3, h = bh & 7;
  const int qrow0 = b * NTOK + blockIdx.x * 128 + wave * 16;   // row into qh/oh

  // persistent q fragments (16 rows x 64 d)
  v16h qa0 = load_a_frag(qh, qrow0, h * HD, CDIM);
  v16h qa1 = load_a_frag(qh, qrow0, h * HD + 32, CDIM);

  v8f o[4] = {};
  float m_run[8], l_run[8];
#pragma unroll
  for (int r = 0; r < 8; ++r) { m_run[r] = -3.0e38f; l_run[r] = 0.f; }

  _Float16* wP = Pl + wave * 16 * 72;

  for (int c = 0; c < NP / 64; ++c) {
    const int kc0 = c * 64;
    // ---- cooperative async stage of K chunk (64x64) and V chunk (64x64) ---
    // 512 x 16B units each, 256 threads -> 2 units per thread per matrix
#pragma unroll
    for (int it = 0; it < 2; ++it) {
      int u = it * 256 + tid;          // 16B unit id
      int row = u >> 3, un = (u & 7) * 8;
      async_copy_16B(kvh + (size_t)(b * NP + kc0 + row) * 1024 + h * HD + un,
                     Kb + row * 64 + un);
      async_copy_16B(vT + (size_t)(b * CDIM + h * HD + row) * 1024 + kc0 + un,
                     Vb + row * 64 + un);
    }
    wait_async_lds();
    __syncthreads();

    // ---- S = q . k^T (4 n-tiles of 16 keys, K = 64 in two steps) ----------
    v8f s[4] = {};
#pragma unroll
    for (int nt = 0; nt < 4; ++nt) {
      v16h kb0 = load_b_frag(Kb, nt * 16, 0, 64);
      v16h kb1 = load_b_frag(Kb, nt * 16, 32, 64);
      s[nt] = WMMA_F16(qa0, kb0, s[nt]);
      s[nt] = WMMA_F16(qa1, kb1, s[nt]);
    }
    // ---- online softmax per row (row r lives in vgpr r across a half) -----
#pragma unroll
    for (int r = 0; r < 8; ++r) {
      float v0 = s[0][r] * ATT_SCALE, v1 = s[1][r] * ATT_SCALE;
      float v2 = s[2][r] * ATT_SCALE, v3 = s[3][r] * ATT_SCALE;
      float mloc = fmaxf(fmaxf(v0, v1), fmaxf(v2, v3));
#pragma unroll
      for (int m = 1; m < 16; m <<= 1) mloc = fmaxf(mloc, __shfl_xor(mloc, m, 16));
      float mnew = fmaxf(m_run[r], mloc);
      float corr = __expf(m_run[r] - mnew);
      float p0 = __expf(v0 - mnew), p1 = __expf(v1 - mnew);
      float p2 = __expf(v2 - mnew), p3 = __expf(v3 - mnew);
      float rsum = p0 + p1 + p2 + p3;
#pragma unroll
      for (int m = 1; m < 16; m <<= 1) rsum += __shfl_xor(rsum, m, 16);
      l_run[r] = l_run[r] * corr + rsum;
      m_run[r] = mnew;
#pragma unroll
      for (int ot = 0; ot < 4; ++ot) o[ot][r] *= corr;
      _Float16* wp = wP + (r + 8 * half) * 72 + cn;
      wp[0] = (_Float16)p0; wp[16] = (_Float16)p1;
      wp[32] = (_Float16)p2; wp[48] = (_Float16)p3;
    }
    __syncthreads();
    // ---- O += P . V (re-fragment P from LDS as A operand) -----------------
    v16h pa0 = load_a_frag(wP, 0, 0, 72);
    v16h pa1 = load_a_frag(wP, 0, 32, 72);
#pragma unroll
    for (int ot = 0; ot < 4; ++ot) {
      v16h vb0 = load_b_frag(Vb, ot * 16, 0, 64);
      v16h vb1 = load_b_frag(Vb, ot * 16, 32, 64);
      o[ot] = WMMA_F16(pa0, vb0, o[ot]);
      o[ot] = WMMA_F16(pa1, vb1, o[ot]);
    }
    __syncthreads();   // protect Kb/Vb/Pl before next chunk's stage
  }
  // normalize and write to oh[b*N + n][h*64 + d] (f16 for proj GEMM)
#pragma unroll
  for (int r = 0; r < 8; ++r) {
    float inv = 1.0f / l_run[r];
    size_t row = (size_t)(qrow0 + r + 8 * half);
#pragma unroll
    for (int ot = 0; ot < 4; ++ot)
      oh[row * CDIM + h * HD + ot * 16 + cn] = (_Float16)(o[ot][r] * inv);
  }
}

// ---------------------------------------------------------------------------
extern "C" void kernel_launch(void* const* d_in, const int* in_sizes, int n_in,
                              void* d_out, int out_size, void* d_ws, size_t ws_size,
                              hipStream_t stream) {
  const float* x      = (const float*)d_in[0];
  const float* Wq     = (const float*)d_in[1];
  const float* Wkv    = (const float*)d_in[2];
  const float* srw    = (const float*)d_in[3];
  const float* srb    = (const float*)d_in[4];
  const float* lng    = (const float*)d_in[5];
  const float* lnb    = (const float*)d_in[6];
  const float* projw  = (const float*)d_in[7];
  const float* projb  = (const float*)d_in[8];
  float* out = (float*)d_out;

  char* w = (char*)d_ws;
  _Float16* xh     = (_Float16*)w;                 w += (size_t)BN * NTOK * CDIM * 2;      // 16.8 MB
  _Float16* Wq_h   = (_Float16*)w;                 w += (size_t)CDIM * CDIM * 2;
  _Float16* Wkv_h  = (_Float16*)w;                 w += (size_t)2 * CDIM * CDIM * 2;
  _Float16* proj_h = (_Float16*)w;                 w += (size_t)CDIM * CDIM * 2;
  _Float16* Wc_h   = (_Float16*)w;                 w += (size_t)CDIM * KCONV * 2;
  _Float16* qhh    = (_Float16*)w;                 w += (size_t)BN * NTOK * CDIM * 2;
  float*    convo  = (float*)w;                    w += (size_t)BN * NP * CDIM * 4;
  _Float16* xr     = (_Float16*)w;                 w += (size_t)BN * NP * CDIM * 2;
  _Float16* kvh    = (_Float16*)w;                 w += (size_t)BN * NP * 2 * CDIM * 2;
  _Float16* vT     = (_Float16*)w;                 w += (size_t)BN * CDIM * NP * 2;
  _Float16* oh     = (_Float16*)w;                 w += (size_t)BN * NTOK * CDIM * 2;

  // 1) precision conversion / packing
  {
    int n = BN * NTOK * CDIM;
    k_f32_to_f16<<<(n + 255) / 256, 256, 0, stream>>>(x, xh, n);
    n = CDIM * CDIM;
    k_f32_to_f16<<<(n + 255) / 256, 256, 0, stream>>>(Wq, Wq_h, n);
    n = 2 * CDIM * CDIM;
    k_f32_to_f16<<<(n + 255) / 256, 256, 0, stream>>>(Wkv, Wkv_h, n);
    n = CDIM * CDIM;
    k_f32_to_f16<<<(n + 255) / 256, 256, 0, stream>>>(projw, proj_h, n);
    n = CDIM * KCONV;
    k_pack_srw<<<(n + 255) / 256, 256, 0, stream>>>(srw, Wc_h);
  }
  // 2) q = x Wq^T  (M=16384, K=512, N=512)
  k_gemm_f16out<<<dim3(CDIM / 64, BN * NTOK / 128), 256, 0, stream>>>(xh, Wq_h, qhh, CDIM, CDIM);
  // 3) spatial-reduction conv as GEMM (M=4096, K=2048, N=512) + bias
  k_conv_gemm<<<dim3(CDIM / 64, BN * NP / 128), 256, 0, stream>>>(xh, Wc_h, srb, convo);
  // 4) LayerNorm -> f16 reduced tokens
  k_layernorm<<<BN * NP / 8, 256, 0, stream>>>(convo, lng, lnb, xr);
  // 5) kv = x_ Wkv^T  (M=4096, K=512, N=1024)
  k_gemm_f16out<<<dim3(2 * CDIM / 64, BN * NP / 128), 256, 0, stream>>>(xr, Wkv_h, kvh, CDIM, 2 * CDIM);
  // 6) transpose v for PV matmul operand layout
  {
    int n = BN * CDIM * NP;
    k_transpose_v<<<(n + 255) / 256, 256, 0, stream>>>(kvh, vT);
  }
  // 7) flash attention (grid: 32 q-tiles x 32 (b,h))
  k_flash_attn<<<dim3(NTOK / 128, BN * HN), 256, 0, stream>>>(qhh, kvh, vT, oh);
  // 8) final projection with bias -> f32 output
  k_gemm_f32out_bias<<<dim3(CDIM / 64, BN * NTOK / 128), 256, 0, stream>>>(oh, proj_h, projb, out, CDIM, CDIM);
}
